// MultiHeadedSelfAttention_11536282157607
// MI455X (gfx1250) — matrix operand order repaired and checked by reading, not compiled, not run on hardware
//
#include <hip/hip_runtime.h>
#include <stdint.h>

typedef __attribute__((ext_vector_type(16))) __bf16          v16bf;
typedef __attribute__((ext_vector_type(8)))  float           v8f;
typedef __attribute__((ext_vector_type(4)))  unsigned int    u32x4;
typedef __attribute__((ext_vector_type(8)))  int             i32x8;
typedef __attribute__((ext_vector_type(4)))  int             i32x4;
typedef __attribute__((ext_vector_type(4)))  float           f32x4;
typedef __attribute__((ext_vector_type(4)))  unsigned short  u16x4;

#define B_  4
#define S_  2048
#define D_  1024
#define H_  16
#define HD_ 64

#if __has_builtin(__builtin_amdgcn_tensor_load_to_lds)
#define HAVE_TDM 1
#endif

__device__ __forceinline__ unsigned short f32_bf16(float f) {
  unsigned int u = __float_as_uint(f);
  u += 0x7FFFu + ((u >> 16) & 1u);          // round-to-nearest-even
  return (unsigned short)(u >> 16);
}

union V16U { u32x4 q[2]; v16bf v; };

__device__ __forceinline__ unsigned lds_byte_off(const void* p) {
  return (unsigned)(uintptr_t)p;            // low 32 bits of generic addr = LDS offset
}

#ifdef HAVE_TDM
// 2D TDM load: tile (tile_d1 rows x tile_d0 elems) of bf16, row stride stride0
// elems in global; LDS rows padded by (pad_amount+1) dwords every 2<<pad_il dwords.
__device__ __forceinline__ void tdm_load_2d(unsigned lds_off, const void* gp,
                                            unsigned tile_d0, unsigned tile_d1,
                                            unsigned stride0,
                                            unsigned pad_il, unsigned pad_am)
{
  const unsigned long long ga = (unsigned long long)(uintptr_t)gp;
  u32x4 g0;
  g0.x = 0x1u;                                            // count=1, user desc
  g0.y = lds_off;                                         // lds_addr
  g0.z = (unsigned)ga;                                    // global_addr[31:0]
  g0.w = (unsigned)((ga >> 32) & 0x1FFFFFFu) | (2u << 30);// addr[56:32] | type=2
  i32x8 g1;
  g1[0] = (int)((1u << 16) | (1u << 20) | (pad_il << 22) | (pad_am << 25)); // 2B, pad_en
  g1[1] = 0;                                              // tensor_dim0[15:0]=0 (2^30)
  g1[2] = 0x4000;                                         // tensor_dim0=2^30, dim1 lo=0
  g1[3] = (int)(0x4000u | (tile_d0 << 16));               // tensor_dim1=2^30 | tile_dim0
  g1[4] = (int)(tile_d1 & 0xFFFFu);                       // tile_dim1 (tile_dim2=0)
  g1[5] = (int)stride0;                                   // tensor_dim0_stride
  g1[6] = 0; g1[7] = 0;
  i32x4 z4 = {0, 0, 0, 0};
#if __clang_major__ >= 23
  i32x8 z8 = {0, 0, 0, 0, 0, 0, 0, 0};
  __builtin_amdgcn_tensor_load_to_lds(g0, g1, z4, z4, z8, 0);
#else
  __builtin_amdgcn_tensor_load_to_lds(g0, g1, z4, z4, 0);
#endif
}
#endif

// ---------------------------------------------------------------------------
// One-shot converters (bandwidth-bound pre-passes).
// ---------------------------------------------------------------------------
__global__ __launch_bounds__(256)
void cvt_bf16(const float* __restrict__ in, unsigned short* __restrict__ out) {
  const size_t i = ((size_t)blockIdx.x * 256 + threadIdx.x) * 8;
  f32x4 a = *(const f32x4*)(in + i);
  f32x4 b = *(const f32x4*)(in + i + 4);
  u16x4 lo = { f32_bf16(a.x), f32_bf16(a.y), f32_bf16(a.z), f32_bf16(a.w) };
  u16x4 hi = { f32_bf16(b.x), f32_bf16(b.y), f32_bf16(b.z), f32_bf16(b.w) };
  *(u16x4*)(out + i)     = lo;
  *(u16x4*)(out + i + 4) = hi;
}

// W:(n,k) fp32 -> Wt:(k,n) bf16 (transpose so GEMM B tiles are contiguous)
__global__ __launch_bounds__(256)
void cvt_w_t(const float* __restrict__ W, unsigned short* __restrict__ Wt) {
  const int idx  = blockIdx.x * 256 + threadIdx.x;
  const int nrow = idx >> 8;
  const int k0   = (idx & 255) * 4;
  f32x4 v = *(const f32x4*)(W + (size_t)nrow * D_ + k0);
  Wt[(size_t)(k0 + 0) * D_ + nrow] = f32_bf16(v.x);
  Wt[(size_t)(k0 + 1) * D_ + nrow] = f32_bf16(v.y);
  Wt[(size_t)(k0 + 2) * D_ + nrow] = f32_bf16(v.z);
  Wt[(size_t)(k0 + 3) * D_ + nrow] = f32_bf16(v.w);
}

// ---------------------------------------------------------------------------
// GEMM: Y = X @ W^T + bias, all-bf16 inputs: X:(8192,1024) row-major,
// Wt:(1024,1024) [k][n]. Block = 8 waves, tile 32(M) x 256(N), K step 32.
// Tiles DMA'd to LDS by the Tensor Data Mover, double-buffered.
// mode 0=Q(scale 1/8, bf16 BHSD) 1=K(bf16 BHDS) 2=V(bf16 BHSD) 3=O(fp32 BSD)
// ---------------------------------------------------------------------------
#define PA 40    // LDS A row stride (elems): 32 data + 8 pad (16+4 dwords)
#define PB 264   // LDS B row stride (elems): 256 data + 8 pad (128+4 dwords)

__global__ __launch_bounds__(256)
void gemm_wmma(const unsigned short* __restrict__ X,
               const unsigned short* __restrict__ Wt,
               const float* __restrict__ bias,
               void* __restrict__ out, int mode)
{
  enum { A0 = 0, A1 = 32 * PA, B0 = 2 * 32 * PA, B1 = B0 + 32 * PB,
         TOT = B1 + 32 * PB };
  __shared__ unsigned short lds[TOT];

  const int tid  = threadIdx.x;
  const int wave = tid >> 5, lane = tid & 31;
  const int bm = blockIdx.x >> 2, bn = blockIdx.x & 3;
  const int row0 = bm * 32, n0 = bn * 256;
  const int ms = (wave & 1) * 16, ns = (wave >> 1) * 64;
  const int hlf = lane >> 4, nl = lane & 15;
  const int chunk = hlf * 8;

  v8f acc0 = {}, acc1 = {}, acc2 = {}, acc3 = {};

#ifdef HAVE_TDM
  const unsigned aoff[2] = { lds_byte_off(&lds[A0]), lds_byte_off(&lds[A1]) };
  const unsigned boff[2] = { lds_byte_off(&lds[B0]), lds_byte_off(&lds[B1]) };
  if (wave == 0) {
    tdm_load_2d(aoff[0], X  + (size_t)row0 * D_,  32, 32, D_, 3, 3);
    tdm_load_2d(boff[0], Wt + n0,                256, 32, D_, 6, 3);
    __builtin_amdgcn_s_wait_tensorcnt(0);
  }
  __syncthreads();
#endif

  int buf = 0;
  for (int k0 = 0; k0 < D_; k0 += 32) {
#ifdef HAVE_TDM
    const int kn = k0 + 32;
    if (wave == 0 && kn < D_) {            // kick next tile while computing
      tdm_load_2d(aoff[buf ^ 1], X  + (size_t)row0 * D_ + kn,   32, 32, D_, 3, 3);
      tdm_load_2d(boff[buf ^ 1], Wt + (size_t)kn * D_ + n0,    256, 32, D_, 6, 3);
    }
    const unsigned short* la = &lds[buf ? A1 : A0];
    const unsigned short* lb = &lds[buf ? B1 : B0];
#else
    { // manual staging fallback (single buffer)
      const int m = tid >> 3, kk = (tid & 7) * 4;
      *(u16x4*)&lds[A0 + m * PA + kk] =
          *(const u16x4*)(X + (size_t)(row0 + m) * D_ + k0 + kk);
      const int kr = tid & 31, nc = (tid >> 5) * 32;
      const unsigned short* gp = Wt + (size_t)(k0 + kr) * D_ + n0 + nc;
      unsigned short* lp = &lds[B0 + kr * PB + nc];
      *(u32x4*)(lp + 0)  = *(const u32x4*)(gp + 0);
      *(u32x4*)(lp + 8)  = *(const u32x4*)(gp + 8);
      *(u32x4*)(lp + 16) = *(const u32x4*)(gp + 16);
      *(u32x4*)(lp + 24) = *(const u32x4*)(gp + 24);
    }
    __syncthreads();
    const unsigned short* la = &lds[A0];
    const unsigned short* lb = &lds[B0];
#endif

    // A operand (16x32): lane = M (mod 16), reg chunks K {c..c+7, 16+c..16+c+7}
    V16U a;
    a.q[0] = *(const u32x4*)&la[(ms + nl) * PA + chunk];
    a.q[1] = *(const u32x4*)&la[(ms + nl) * PA + 16 + chunk];

    // B operands (32x16): lane = N column, reg elements K {c..c+7, 16+c..16+c+7}
    V16U b0, b1, b2, b3;
    #pragma unroll
    for (int s = 0; s < 16; ++s) {
      const int kr = (s < 8) ? chunk + s : 16 + chunk + (s - 8);
      const unsigned short* bp = &lb[kr * PB + ns + nl];
      b0.v[s] = __builtin_bit_cast(__bf16, bp[0]);
      b1.v[s] = __builtin_bit_cast(__bf16, bp[16]);
      b2.v[s] = __builtin_bit_cast(__bf16, bp[32]);
      b3.v[s] = __builtin_bit_cast(__bf16, bp[48]);
    }

    acc0 = __builtin_amdgcn_wmma_f32_16x16x32_bf16(false, a.v, false, b0.v, (short)0, acc0, false, false);
    acc1 = __builtin_amdgcn_wmma_f32_16x16x32_bf16(false, a.v, false, b1.v, (short)0, acc1, false, false);
    acc2 = __builtin_amdgcn_wmma_f32_16x16x32_bf16(false, a.v, false, b2.v, (short)0, acc2, false, false);
    acc3 = __builtin_amdgcn_wmma_f32_16x16x32_bf16(false, a.v, false, b3.v, (short)0, acc3, false, false);

#ifdef HAVE_TDM
    if (wave == 0 && kn < D_) __builtin_amdgcn_s_wait_tensorcnt(0);
    __syncthreads();
    buf ^= 1;
#else
    __syncthreads();
#endif
  }

  // ---- epilogue: bias + layout-specific store ----
  v8f accs[4] = {acc0, acc1, acc2, acc3};
  #pragma unroll
  for (int t = 0; t < 4; ++t) {
    #pragma unroll
    for (int r = 0; r < 8; ++r) {
      const int gm = row0 + ms + hlf * 8 + r;        // global row (b*S+s)
      const int gn = n0 + ns + t * 16 + nl;          // global out-dim (h*HD+hd)
      float val = accs[t][r] + bias[gn];
      const int bb = gm >> 11, ss = gm & 2047;
      const int hh = gn >> 6,  hd = gn & 63;
      if (mode == 0) {
        ((unsigned short*)out)[((size_t)(bb * H_ + hh) * S_ + ss) * HD_ + hd] =
            f32_bf16(val * 0.125f);                   // fold 1/sqrt(64)
      } else if (mode == 1) {
        ((unsigned short*)out)[((size_t)(bb * H_ + hh) * HD_ + hd) * S_ + ss] =
            f32_bf16(val);                            // K transposed
      } else if (mode == 2) {
        ((unsigned short*)out)[((size_t)(bb * H_ + hh) * S_ + ss) * HD_ + hd] =
            f32_bf16(val);
      } else {
        ((float*)out)[(size_t)gm * D_ + gn] = val;
      }
    }
  }
}

// ---------------------------------------------------------------------------
// Causal flash attention. Block = 256 thr = 8 waves; wave owns 16 query rows.
// Q:(B,H,S,HD) bf16 (pre-scaled), Kt:(B,H,HD,S) bf16, V:(B,H,S,HD) bf16.
// Ctx out: (B,S,D) bf16.
// ---------------------------------------------------------------------------
#define PP 40    // per-wave LDS P-tile row stride (elements)

__global__ __launch_bounds__(256)
void attn_flash(const unsigned short* __restrict__ Q,
                const unsigned short* __restrict__ Kt,
                const unsigned short* __restrict__ V,
                unsigned short* __restrict__ Ctx)
{
  __shared__ unsigned short ldsP[8 * 16 * PP];

  const int tid  = threadIdx.x;
  const int wave = tid >> 5, lane = tid & 31;
  const int hlf = lane >> 4, nl = lane & 15;
  const int chunk = hlf * 8;

  const int qblk = blockIdx.x & 15;
  const int bh   = blockIdx.x >> 4;        // b*H + h
  const int b    = bh >> 4;
  const int h    = bh & 15;
  const int qbase = qblk * 128 + wave * 16;

  const unsigned short* qp  = Q  + (size_t)bh * S_ * HD_;
  const unsigned short* ktp = Kt + (size_t)bh * HD_ * S_;
  const unsigned short* vp  = V  + (size_t)bh * S_ * HD_;
  unsigned short* pw = &ldsP[wave * 16 * PP];

  // Q A-operands (hd 0..31 and 32..63), loaded once per wave
  const unsigned short* qr = qp + (size_t)(qbase + nl) * HD_;
  V16U a0, a1;
  a0.q[0] = *(const u32x4*)(qr + chunk);       a0.q[1] = *(const u32x4*)(qr + 16 + chunk);
  a1.q[0] = *(const u32x4*)(qr + 32 + chunk);  a1.q[1] = *(const u32x4*)(qr + 48 + chunk);

  v8f o0 = {}, o1 = {}, o2 = {}, o3 = {};
  float mrow[8], lrow[8];
  #pragma unroll
  for (int r = 0; r < 8; ++r) { mrow[r] = -1e30f; lrow[r] = 0.0f; }

  const int qend = qbase + 15;
  for (int kb = 0; kb <= qend; kb += 32) {
    // ---- scores S = Q @ K^T for 32 keys (two 16-key tiles) ----
    V16U bk00, bk01, bk10, bk11;
    #pragma unroll
    for (int s = 0; s < 16; ++s) {
      const int kr = (s < 8) ? chunk + s : 16 + chunk + (s - 8);   // hd
      const unsigned short* k0p = ktp + (size_t)kr * S_ + kb + nl;
      const unsigned short* k1p = ktp + (size_t)(32 + kr) * S_ + kb + nl;
      bk00.v[s] = __builtin_bit_cast(__bf16, k0p[0]);
      bk01.v[s] = __builtin_bit_cast(__bf16, k0p[16]);
      bk10.v[s] = __builtin_bit_cast(__bf16, k1p[0]);
      bk11.v[s] = __builtin_bit_cast(__bf16, k1p[16]);
    }

    v8f s0 = {}, s1 = {};
    s0 = __builtin_amdgcn_wmma_f32_16x16x32_bf16(false, a0.v, false, bk00.v, (short)0, s0, false, false);
    s0 = __builtin_amdgcn_wmma_f32_16x16x32_bf16(false, a1.v, false, bk10.v, (short)0, s0, false, false);
    s1 = __builtin_amdgcn_wmma_f32_16x16x32_bf16(false, a0.v, false, bk01.v, (short)0, s1, false, false);
    s1 = __builtin_amdgcn_wmma_f32_16x16x32_bf16(false, a1.v, false, bk11.v, (short)0, s1, false, false);

    // ---- causal mask + online softmax (C layout: row = r + 8*hlf) ----
    #pragma unroll
    for (int r = 0; r < 8; ++r) {
      const int qg = qbase + r + hlf * 8;
      float sv0 = s0[r], sv1 = s1[r];
      if (kb + nl > qg)      sv0 = -1e30f;
      if (kb + 16 + nl > qg) sv1 = -1e30f;
      float mt = fmaxf(sv0, sv1);
      #pragma unroll
      for (int off = 1; off < 16; off <<= 1) mt = fmaxf(mt, __shfl_xor(mt, off, 32));
      const float mnew = fmaxf(mrow[r], mt);
      const float p0 = __expf(sv0 - mnew);
      const float p1 = __expf(sv1 - mnew);
      const float corr = __expf(mrow[r] - mnew);
      mrow[r] = mnew;
      float rs = p0 + p1;
      #pragma unroll
      for (int off = 1; off < 16; off <<= 1) rs += __shfl_xor(rs, off, 32);
      lrow[r] = lrow[r] * corr + rs;
      o0[r] *= corr; o1[r] *= corr; o2[r] *= corr; o3[r] *= corr;
      const int m = r + hlf * 8;
      pw[m * PP + nl]      = f32_bf16(p0);   // transpose C->A via LDS
      pw[m * PP + 16 + nl] = f32_bf16(p1);
    }

    // ---- reload P in A layout (16 queries x 32 keys) ----
    V16U pa;
    pa.q[0] = *(const u32x4*)&pw[nl * PP + chunk];
    pa.q[1] = *(const u32x4*)&pw[nl * PP + 16 + chunk];

    // ---- O += P @ V : B column = hd (lane), elements = keys, 4 hd tiles of 16 ----
    V16U vb0, vb1, vb2, vb3;
    #pragma unroll
    for (int s = 0; s < 16; ++s) {
      const int kr = (s < 8) ? chunk + s : 16 + chunk + (s - 8);   // key
      const unsigned short* vrow = vp + (size_t)(kb + kr) * HD_ + nl;
      vb0.v[s] = __builtin_bit_cast(__bf16, vrow[0]);
      vb1.v[s] = __builtin_bit_cast(__bf16, vrow[16]);
      vb2.v[s] = __builtin_bit_cast(__bf16, vrow[32]);
      vb3.v[s] = __builtin_bit_cast(__bf16, vrow[48]);
    }

    o0 = __builtin_amdgcn_wmma_f32_16x16x32_bf16(false, pa.v, false, vb0.v, (short)0, o0, false, false);
    o1 = __builtin_amdgcn_wmma_f32_16x16x32_bf16(false, pa.v, false, vb1.v, (short)0, o1, false, false);
    o2 = __builtin_amdgcn_wmma_f32_16x16x32_bf16(false, pa.v, false, vb2.v, (short)0, o2, false, false);
    o3 = __builtin_amdgcn_wmma_f32_16x16x32_bf16(false, pa.v, false, vb3.v, (short)0, o3, false, false);
  }

  // ---- normalize and store context bf16 in (B,S,D) ----
  #pragma unroll
  for (int r = 0; r < 8; ++r) lrow[r] = 1.0f / lrow[r];
  #pragma unroll
  for (int r = 0; r < 8; ++r) {
    const int sg = qbase + r + hlf * 8;
    const size_t base = ((size_t)b * S_ + sg) * D_ + h * HD_;
    Ctx[base +  0 + nl] = f32_bf16(o0[r] * lrow[r]);
    Ctx[base + 16 + nl] = f32_bf16(o1[r] * lrow[r]);
    Ctx[base + 32 + nl] = f32_bf16(o2[r] * lrow[r]);
    Ctx[base + 48 + nl] = f32_bf16(o3[r] * lrow[r]);
  }
}

// ---------------------------------------------------------------------------
extern "C" void kernel_launch(void* const* d_in, const int* in_sizes, int n_in,
                              void* d_out, int out_size, void* d_ws, size_t ws_size,
                              hipStream_t stream)
{
  const float* x  = (const float*)d_in[0];
  const float* Wq = (const float*)d_in[1];
  const float* bq = (const float*)d_in[2];
  const float* Wk = (const float*)d_in[3];
  const float* bk = (const float*)d_in[4];
  const float* Wv = (const float*)d_in[5];
  const float* bv = (const float*)d_in[6];
  const float* Wo = (const float*)d_in[7];
  const float* bo = (const float*)d_in[8];

  const size_t nX = (size_t)B_ * S_ * D_;          // 8M elements
  const size_t nW = (size_t)D_ * D_;               // 1M elements
  unsigned short* xb   = (unsigned short*)d_ws;    // bf16 x          (16MB)
  unsigned short* wtq  = xb   + nX;                // bf16 Wq^T       (2MB)
  unsigned short* wtk  = wtq  + nW;
  unsigned short* wtv  = wtk  + nW;
  unsigned short* wto  = wtv  + nW;
  unsigned short* qws  = wto  + nW;                // Q  (B,H,S,HD)   (16MB)
  unsigned short* ktws = qws  + nX;                // Kt (B,H,HD,S)
  unsigned short* vws  = ktws + nX;                // V  (B,H,S,HD)
  unsigned short* ctx  = vws  + nX;                // ctx (B,S,D)

  const dim3 blk(256);
  cvt_bf16<<<dim3(nX / 2048), blk, 0, stream>>>(x, xb);
  cvt_w_t<<<dim3(nW / 1024), blk, 0, stream>>>(Wq, wtq);
  cvt_w_t<<<dim3(nW / 1024), blk, 0, stream>>>(Wk, wtk);
  cvt_w_t<<<dim3(nW / 1024), blk, 0, stream>>>(Wv, wtv);
  cvt_w_t<<<dim3(nW / 1024), blk, 0, stream>>>(Wo, wto);

  const dim3 gg(256 * 4);
  gemm_wmma<<<gg, blk, 0, stream>>>(xb, wtq, bq, qws,  0);
  gemm_wmma<<<gg, blk, 0, stream>>>(xb, wtk, bk, ktws, 1);
  gemm_wmma<<<gg, blk, 0, stream>>>(xb, wtv, bv, vws,  2);
  attn_flash<<<dim3(B_ * H_ * 16), blk, 0, stream>>>(qws, ktws, vws, ctx);
  gemm_wmma<<<gg, blk, 0, stream>>>(ctx, wto, bo, (float*)d_out, 3);
}
